// RNN_69157563400379
// MI455X (gfx1250) — matrix operand order repaired and checked by reading, not compiled, not run on hardware
//
#include <hip/hip_runtime.h>
#include <hip/hip_bf16.h>

typedef __attribute__((ext_vector_type(16))) __bf16 v16bf;
typedef __attribute__((ext_vector_type(8)))  float  v8f;

#define HDIM 768
#define SLEN 1024
#define BDIM 64
#define LDSPITCH 776   // 768 + 8 bf16 pad -> 1552B row pitch, conflict-free ds_load_b128

union FragU { uint4 q[2]; v16bf v; };

// 16-bit A-matrix 16x32 layout: lane<16 holds K {0..7,16..23}, lane>=16 holds K {8..15,24..31}
__device__ __forceinline__ v16bf load_frag_a(const __hip_bfloat16* rowPtr, int k0, int hi) {
    FragU f;
    f.q[0] = *reinterpret_cast<const uint4*>(rowPtr + k0 + hi * 8);
    f.q[1] = *reinterpret_cast<const uint4*>(rowPtr + k0 + hi * 8 + 16);
    return f.v;
}

// B-matrix 32x16: lane&15 = column, lane<16 holds K {0..7,16..23}, lane>=16 holds K {8..15,24..31}
__device__ __forceinline__ v16bf load_frag_b(const __hip_bfloat16* rowPtr, int k0, int hi) {
    const uint4* p = reinterpret_cast<const uint4*>(rowPtr + k0 + hi * 8);
    FragU f;
    f.q[0] = p[0];
    f.q[1] = p[2];
    return f.v;
}

// ---------------- fp32 -> bf16 conversion ----------------
__global__ void cvt_f32_bf16(const float* __restrict__ src,
                             __hip_bfloat16* __restrict__ dst, long long n) {
    long long i = (long long)blockIdx.x * blockDim.x + threadIdx.x;
    if (i < n) dst[i] = __float2bfloat16(src[i]);
}

__global__ void init_counter(unsigned int* c) { *c = 0u; }

// ---------------- big GEMM: out[M,768] = A[M,768](bf16) @ W[768,768]^T(bf16) + bias ----
// grid = (M/128, 768/64), block = 256 (8 waves). Wave w: m-tile w, 4 n-tiles (A reuse x4).
__global__ __launch_bounds__(256) void gemm_bf16_768(
    const __hip_bfloat16* __restrict__ A,
    const __hip_bfloat16* __restrict__ W,
    const float* __restrict__ bias0,
    const float* __restrict__ bias1,
    float* __restrict__ out, int M)
{
    const int lane = threadIdx.x & 31;
    const int wave = threadIdx.x >> 5;
    const int hi   = lane >> 4;
    const int l15  = lane & 15;
    const int mBase = blockIdx.x * 128 + wave * 16;
    const int nBase = blockIdx.y * 64;

    const __hip_bfloat16* aRow = A + (size_t)(mBase + l15) * HDIM;
    v8f acc[4] = {};

#pragma unroll 4
    for (int kt = 0; kt < HDIM / 32; ++kt) {
        const int k0 = kt * 32;
        v16bf aF = load_frag_a(aRow, k0, hi);
#pragma unroll
        for (int j = 0; j < 4; ++j) {
            const __hip_bfloat16* bRow = W + (size_t)(nBase + j * 16 + l15) * HDIM;
            v16bf bF = load_frag_b(bRow, k0, hi);
            acc[j] = __builtin_amdgcn_wmma_f32_16x16x32_bf16(
                false, aF, false, bF, (short)0, acc[j], false, false);
        }
    }

#pragma unroll
    for (int j = 0; j < 4; ++j) {
        const int n = nBase + j * 16 + l15;
        const float bsum = bias0[n] + (bias1 ? bias1[n] : 0.0f);
#pragma unroll
        for (int r = 0; r < 8; ++r) {
            const int row = mBase + r + hi * 8;
            out[(size_t)row * HDIM + n] = acc[j][r] + bsum;
        }
    }
}

// ---------------- persistent recurrent scan ----------------
// grid = 24 WGs (each owns 32 columns of H), block = 256 (8 waves: 4 m-tiles x 2 n-tiles).
// W_hh column-slice lives in LDS (bf16). Hidden state ping-pongs between two global bf16
// buffers; one grid-wide barrier per timestep via release-add / acquire-spin counter.
__global__ __launch_bounds__(256) void rnn_scan(
    const float* __restrict__ W_hh,        // [768,768] fp32
    const float* __restrict__ xp,          // [B*S, H] fp32 (lives in d_out)
    __hip_bfloat16* __restrict__ h_state,  // 2 x [64,768] bf16 ping-pong
    __hip_bfloat16* __restrict__ outs,     // [B*S, H] bf16
    float* __restrict__ hT,                // [64,768] fp32 final hidden
    unsigned int* __restrict__ counter)
{
    __shared__ __align__(16) __hip_bfloat16 wlds[32 * LDSPITCH];

    const int tid  = threadIdx.x;
    const int lane = tid & 31;
    const int wave = tid >> 5;
    const int hi   = lane >> 4;
    const int l15  = lane & 15;
    const int nSlice = blockIdx.x * 32;

    // stage W_hh[nSlice .. nSlice+31, :] into LDS as bf16
    for (int idx = tid; idx < 32 * HDIM; idx += 256) {
        const int j = idx / HDIM, k = idx - j * HDIM;
        wlds[j * LDSPITCH + k] =
            __float2bfloat16(W_hh[(size_t)(nSlice + j) * HDIM + k]);
    }
    __syncthreads();

    const int mTile = wave & 3;          // 4 batch tiles of 16
    const int nLoc  = wave >> 2;         // 2 column tiles of 16
    const int bBase = mTile * 16;
    const int nG    = nSlice + nLoc * 16 + l15;
    const __hip_bfloat16* bRow = wlds + (nLoc * 16 + l15) * LDSPITCH;
    const unsigned nWG = gridDim.x;
    const int HS = BDIM * HDIM;          // elements per state buffer

    for (int t = 0; t < SLEN; ++t) {
        const __hip_bfloat16* hRead = h_state + (t & 1 ? HS : 0);
        __hip_bfloat16*       hWrite = h_state + (t & 1 ? 0 : HS);
        const __hip_bfloat16* aRow = hRead + (size_t)(bBase + l15) * HDIM;

        v8f acc = {};
#pragma unroll 4
        for (int kt = 0; kt < HDIM / 32; ++kt) {
            const int k0 = kt * 32;
            v16bf aF = load_frag_a(aRow, k0, hi);
            v16bf bF = load_frag_b(bRow, k0, hi);
            acc = __builtin_amdgcn_wmma_f32_16x16x32_bf16(
                false, aF, false, bF, (short)0, acc, false, false);
        }

#pragma unroll
        for (int r = 0; r < 8; ++r) {
            const int b = bBase + r + hi * 8;
            const size_t rowIdx = (size_t)b * SLEN + t;     // row in [B*S,H]
            const float v = tanhf(acc[r] + xp[rowIdx * HDIM + nG]);
            outs[rowIdx * HDIM + nG]      = __float2bfloat16(v);
            hWrite[(size_t)b * HDIM + nG] = __float2bfloat16(v);
            if (t == SLEN - 1) hT[(size_t)b * HDIM + nG] = v;
        }

        // grid-wide barrier: all 24 WGs finish step t before anyone reads hWrite
        __threadfence();
        __syncthreads();
        if (tid == 0) {
            __hip_atomic_fetch_add(counter, 1u, __ATOMIC_ACQ_REL,
                                   __HIP_MEMORY_SCOPE_AGENT);
            const unsigned target = nWG * (unsigned)(t + 1);
            while (__hip_atomic_load(counter, __ATOMIC_ACQUIRE,
                                     __HIP_MEMORY_SCOPE_AGENT) < target)
                __builtin_amdgcn_s_sleep(1);
        }
        __syncthreads();
    }
}

extern "C" void kernel_launch(void* const* d_in, const int* in_sizes, int n_in,
                              void* d_out, int out_size, void* d_ws, size_t ws_size,
                              hipStream_t stream) {
    const float* hidden = (const float*)d_in[0];   // [1,64,768]
    const float* x      = (const float*)d_in[1];   // [64,1024,768]
    const float* W_ih   = (const float*)d_in[2];   // [768,768]
    const float* W_hh   = (const float*)d_in[3];   // [768,768]
    const float* b_ih   = (const float*)d_in[4];   // [768]
    const float* b_hh   = (const float*)d_in[5];   // [768]
    const float* W_fc   = (const float*)d_in[6];   // [768,768]
    const float* b_fc   = (const float*)d_in[7];   // [768]
    float* out = (float*)d_out;                    // [65536,768] logits ++ [64,768] hT

    const long long NX = (long long)BDIM * SLEN * HDIM;   // 50,331,648
    const long long NW = (long long)HDIM * HDIM;          //    589,824
    const long long NH = (long long)BDIM * HDIM;          //     49,152

    char* ws = (char*)d_ws;
    // [0, 96MiB): x_bf16, later reused as outs_bf16 (identical shape [B*S,H])
    __hip_bfloat16* bufBig = (__hip_bfloat16*)(ws);
    __hip_bfloat16* hstate = (__hip_bfloat16*)(ws + 100663296);   // 2 x [64,768] bf16
    __hip_bfloat16* wihB   = (__hip_bfloat16*)(ws + 100859904);
    __hip_bfloat16* wfcB   = (__hip_bfloat16*)(ws + 102039552);
    unsigned int*   counter = (unsigned int*)(ws + 103219200);

    // 1) fp32 -> bf16 staging
    cvt_f32_bf16<<<(unsigned)((NX + 255) / 256), 256, 0, stream>>>(x, bufBig, NX);
    cvt_f32_bf16<<<(unsigned)((NW + 255) / 256), 256, 0, stream>>>(W_ih, wihB, NW);
    cvt_f32_bf16<<<(unsigned)((NW + 255) / 256), 256, 0, stream>>>(W_fc, wfcB, NW);
    cvt_f32_bf16<<<(unsigned)((NH + 255) / 256), 256, 0, stream>>>(hidden, hstate, NH); // buf 0
    init_counter<<<1, 1, 0, stream>>>(counter);

    // 2) xp = x @ W_ih^T + b_ih + b_hh   -> stored fp32 in d_out logits region
    gemm_bf16_768<<<dim3(512, 12), 256, 0, stream>>>(bufBig, wihB, b_ih, b_hh, out, 65536);

    // 3) persistent recurrent scan: writes outs (bf16, reusing bufBig) + hT tail of d_out
    rnn_scan<<<24, 256, 0, stream>>>(W_hh, out, hstate, bufBig,
                                     out + (size_t)65536 * HDIM, counter);

    // 4) logits = outs @ W_fc^T + b_fc   -> overwrites d_out logits region
    gemm_bf16_768<<<dim3(512, 12), 256, 0, stream>>>(bufBig, wfcB, b_fc, nullptr, out, 65536);
}
